// PointTransformerCls_58007828300562
// MI455X (gfx1250) — matrix operand order repaired and checked by reading, not compiled, not run on hardware
//
#include <hip/hip_runtime.h>
#include <cmath>

// ---------------------------------------------------------------------------
// PointTransformer forward for gfx1250 (MI455X). All matmuls go through
// v_wmma_f32_16x16x32_bf16 (bf16 A/B, f32 accumulate). Irregular ops (FPS,
// kNN select, grouping, softmax) are plain wave32 kernels.
// ---------------------------------------------------------------------------

typedef __bf16 bf16;
typedef bf16  v16bf __attribute__((ext_vector_type(16)));
typedef float v8f   __attribute__((ext_vector_type(8)));

#define EPI_NONE         0
#define EPI_BNRELU       1   // v = relu(v*e0[row] + e1[row])
#define EPI_BIASROW      2   // v = v + e1[row]
#define EPI_BNRELU_POOL  3   // bnrelu, then atomic max into pool[b, row, s]
#define EPI_BIASCOL      4   // v = v + e1[col]
#define EPI_BIASCOL_RELU 5   // v = relu(v + e1[col])
#define EPI_RELU         6

// packed f32x2 -> bf16x2 store (dst 4B aligned)
__device__ __forceinline__ void store_pk2(bf16* dst, float a, float b)
{
    union { bf16 h[2]; unsigned u; } t;
    t.h[0] = (bf16)a; t.h[1] = (bf16)b;
    *(unsigned*)dst = t.u;
}

// ---------------------------------------------------------------------------
// Batched GEMM: C[z] = A[z] (MxK) * B[z] (KxN), f32 in/out, bf16 WMMA compute.
// TA: A element (m,k) at A[k*lda + m]   (q^T q energy)
// TB: B element (k,n) at B[n*ldb + k]   (h @ W^T FC layers)
// NW waves per block; macro tile (16*NW) x 32; each wave computes 16x32
// (two 16x16 accumulators, A-fragment reuse -> 2 WMMAs per k-step).
// Fast path (full tile, K%32==0): vectorized float4 loads, packed bf16 stores.
// ---------------------------------------------------------------------------
template <int TA, int TB, int NW>
__global__ __launch_bounds__(NW * 32)
void k_gemm(const float* __restrict__ A, long lda,
            const float* __restrict__ Bm, long ldb,
            float* __restrict__ C, long ldc,
            int M, int N, int K,
            long sA, long sB, long sC,
            int epi, const float* __restrict__ e0, const float* __restrict__ e1,
            float* __restrict__ pool, int poolN)
{
    constexpr int ROWS_A = 16 * NW;      // macro tile M extent
    constexpr int CH     = 32 / NW;      // B chunk per thread (fast path)
    __shared__ bf16 As[ROWS_A][34];      // [m][k], K-contiguous
    __shared__ bf16 Bs[32][34];          // [n][k], K-contiguous

    const int tid  = threadIdx.x;
    const int z    = blockIdx.z;
    const int m0   = blockIdx.y * ROWS_A;
    const int n0   = blockIdx.x * 32;
    const float* Az = A  + (long)z * sA;
    const float* Bz = Bm + (long)z * sB;

    const int wave = tid >> 5;
    const int lane = tid & 31;
    const int half = lane >> 4;
    const int l16  = lane & 15;
    const int wm   = wave * 16;          // wave's row offset in macro tile

    v8f acc0 = {0.f, 0.f, 0.f, 0.f, 0.f, 0.f, 0.f, 0.f};
    v8f acc1 = {0.f, 0.f, 0.f, 0.f, 0.f, 0.f, 0.f, 0.f};

    const bool full = (m0 + ROWS_A <= M) && (n0 + 32 <= N) && ((K & 31) == 0);

    for (int k0 = 0; k0 < K; k0 += 32) {
        if (full) {
            // ---- A tile ----
            if constexpr (TA == 0) {
                // rows K-contiguous: float4 loads, packed bf16 stores
                const int r  = tid & (ROWS_A - 1);
                const int cb = (tid / ROWS_A) * 16;
                const float* src = Az + (long)(m0 + r) * lda + (k0 + cb);
#pragma unroll
                for (int q = 0; q < 4; ++q) {
                    float4 v4 = *(const float4*)(src + 4 * q);
                    store_pk2(&As[r][cb + 4 * q],     v4.x, v4.y);
                    store_pk2(&As[r][cb + 4 * q + 2], v4.z, v4.w);
                }
            } else {
                // contiguous along m: float4 loads, scattered bf16 stores
                const int k  = tid & 31;
                const int mb = (tid >> 5) * 16;
                const float* src = Az + (long)(k0 + k) * lda + (m0 + mb);
#pragma unroll
                for (int q = 0; q < 4; ++q) {
                    float4 v4 = *(const float4*)(src + 4 * q);
                    As[mb + 4 * q + 0][k] = (bf16)v4.x;
                    As[mb + 4 * q + 1][k] = (bf16)v4.y;
                    As[mb + 4 * q + 2][k] = (bf16)v4.z;
                    As[mb + 4 * q + 3][k] = (bf16)v4.w;
                }
            }
            // ---- B tile ----
            if constexpr (TB == 0) {
                // contiguous along n: float4 loads, transpose-scatter to [n][k]
                const int k  = tid & 31;
                const int nb = (tid >> 5) * CH;
                const float* src = Bz + (long)(k0 + k) * ldb + (n0 + nb);
#pragma unroll
                for (int q = 0; q < CH / 4; ++q) {
                    float4 v4 = *(const float4*)(src + 4 * q);
                    Bs[nb + 4 * q + 0][k] = (bf16)v4.x;
                    Bs[nb + 4 * q + 1][k] = (bf16)v4.y;
                    Bs[nb + 4 * q + 2][k] = (bf16)v4.z;
                    Bs[nb + 4 * q + 3][k] = (bf16)v4.w;
                }
            } else {
                // rows K-contiguous: float4 loads, packed stores
                const int nloc = tid & 31;
                const int kb   = (tid >> 5) * CH;
                const float* src = Bz + (long)(n0 + nloc) * ldb + (k0 + kb);
#pragma unroll
                for (int q = 0; q < CH / 4; ++q) {
                    float4 v4 = *(const float4*)(src + 4 * q);
                    store_pk2(&Bs[nloc][kb + 4 * q],     v4.x, v4.y);
                    store_pk2(&Bs[nloc][kb + 4 * q + 2], v4.z, v4.w);
                }
            }
        } else {
            // guarded scalar path for ragged tiles
            for (int i = tid; i < ROWS_A * 32; i += NW * 32) {
                int r = i >> 5, c = i & 31;
                int gm = m0 + r, gk = k0 + c;
                float va = 0.f;
                if (gm < M && gk < K)
                    va = TA ? Az[(long)gk * lda + gm] : Az[(long)gm * lda + gk];
                As[r][c] = (bf16)va;
            }
            for (int i = tid; i < 32 * 32; i += NW * 32) {
                int r = i >> 5, c = i & 31;
                int gn = n0 + r, gk = k0 + c;
                float vb = 0.f;
                if (gn < N && gk < K)
                    vb = TB ? Bz[(long)gn * ldb + gk] : Bz[(long)gk * ldb + gn];
                Bs[r][c] = (bf16)vb;
            }
        }
        __syncthreads();

        // Build fragments in contiguous k order (k = 16*half + s, same for A and B) and issue 2 WMMAs.
        v16bf fa, fb0, fb1;
#pragma unroll
        for (int p = 0; p < 8; ++p) {
            int ka = 16 * half + 2 * p;
            fa[2 * p]      = As[wm + l16][ka];
            fa[2 * p + 1]  = As[wm + l16][ka + 1];
            int kb = 16 * half + 2 * p;
            fb0[2 * p]     = Bs[l16][kb];
            fb0[2 * p + 1] = Bs[l16][kb + 1];
            fb1[2 * p]     = Bs[16 + l16][kb];
            fb1[2 * p + 1] = Bs[16 + l16][kb + 1];
        }
        acc0 = __builtin_amdgcn_wmma_f32_16x16x32_bf16(false, fa, false, fb0,
                                                       (short)0, acc0, false, false);
        acc1 = __builtin_amdgcn_wmma_f32_16x16x32_bf16(false, fa, false, fb1,
                                                       (short)0, acc1, false, false);
        __syncthreads();
    }

    // C/D layout: lane -> col = lane&15 ; VGPR j -> row = j + 8*(lane>>4)
    const int gn0 = n0 + l16;
    const int gn1 = n0 + 16 + l16;
#pragma unroll
    for (int j = 0; j < 8; ++j) {
        int gm = m0 + wm + 8 * half + j;
        if (gm < M) {
#pragma unroll
            for (int t = 0; t < 2; ++t) {
                int gn = t ? gn1 : gn0;
                if (gn >= N) continue;
                float v = t ? acc1[j] : acc0[j];
                if (epi == EPI_BNRELU_POOL) {
                    v = fmaxf(v * e0[gm] + e1[gm], 0.f);
                    int bidx = z / poolN, s = z % poolN;
                    // post-relu values >= 0: int-bit atomic max exact (pool zeroed)
                    atomicMax((int*)(pool + ((long)bidx * M + gm) * poolN + s),
                              __float_as_int(v));
                } else {
                    if (epi == EPI_BNRELU)            v = fmaxf(v * e0[gm] + e1[gm], 0.f);
                    else if (epi == EPI_BIASROW)      v = v + e1[gm];
                    else if (epi == EPI_BIASCOL)      v = v + e1[gn];
                    else if (epi == EPI_BIASCOL_RELU) v = fmaxf(v + e1[gn], 0.f);
                    else if (epi == EPI_RELU)         v = fmaxf(v, 0.f);
                    C[(long)z * sC + (long)gm * ldc + gn] = v;
                }
            }
        }
    }
}

// ---------------------------------------------------------------------------
// BN parameter folding: scale = g*rsqrt(v+eps), shift = b - m*scale
// ---------------------------------------------------------------------------
__global__ void k_bnprep(const float* __restrict__ g, const float* __restrict__ b,
                         const float* __restrict__ m, const float* __restrict__ v,
                         float* __restrict__ scale, float* __restrict__ shift, int C)
{
    int i = blockIdx.x * 256 + threadIdx.x;
    if (i < C) {
        float inv = g[i] * rsqrtf(v[i] + 1e-5f);
        scale[i] = inv;
        shift[i] = b[i] - m[i] * inv;
    }
}

__global__ void k_zero(float* __restrict__ p, long n)
{
    long i = (long)blockIdx.x * 256 + threadIdx.x;
    if (i < n) p[i] = 0.f;
}

__global__ void k_add3(float* __restrict__ o, const float* __restrict__ a,
                       const float* __restrict__ b, long n)
{
    long i = (long)blockIdx.x * 256 + threadIdx.x;
    if (i < n) o[i] += a[i] + b[i];
}

// ---------------------------------------------------------------------------
// Farthest point sampling: one block per batch, sequential scan with shared
// argmax (first-index tie-break, matching jnp.argmax).
// ---------------------------------------------------------------------------
__global__ __launch_bounds__(256)
void k_fps(const float* __restrict__ xyz, int n, int npoint, int* __restrict__ out)
{
    __shared__ float rd[256];
    __shared__ int   ri[256];
    __shared__ float cen[3];
    __shared__ int   sfar;
    const int b = blockIdx.x, tid = threadIdx.x;
    const float* xb = xyz + (long)b * n * 3;
    float dist[8];
#pragma unroll
    for (int k = 0; k < 8; ++k) dist[k] = 1e10f;
    if (tid == 0) sfar = 0;
    __syncthreads();
    for (int it = 0; it < npoint; ++it) {
        int far = sfar;
        if (tid == 0) {
            out[(long)b * npoint + it] = far;
            cen[0] = xb[far * 3]; cen[1] = xb[far * 3 + 1]; cen[2] = xb[far * 3 + 2];
        }
        __syncthreads();
        float best = -1.f; int bi = n;
#pragma unroll
        for (int k = 0; k < 8; ++k) {
            int p = tid + k * 256;
            if (p < n) {
                float dx = xb[p * 3] - cen[0];
                float dy = xb[p * 3 + 1] - cen[1];
                float dz = xb[p * 3 + 2] - cen[2];
                float dd = dx * dx + dy * dy + dz * dz;
                dist[k] = fminf(dist[k], dd);
                if (dist[k] > best || (dist[k] == best && p < bi)) { best = dist[k]; bi = p; }
            }
        }
        rd[tid] = best; ri[tid] = bi;
        __syncthreads();
        for (int s = 128; s > 0; s >>= 1) {
            if (tid < s) {
                if (rd[tid + s] > rd[tid] ||
                    (rd[tid + s] == rd[tid] && ri[tid + s] < ri[tid])) {
                    rd[tid] = rd[tid + s]; ri[tid] = ri[tid + s];
                }
            }
            __syncthreads();
        }
        if (tid == 0) sfar = ri[0];
        __syncthreads();
    }
}

__global__ void k_gather_xyz(const float* __restrict__ xyz, const int* __restrict__ idx,
                             float* __restrict__ nx, int n, int np, int total)
{
    int t = blockIdx.x * 256 + threadIdx.x;
    if (t < total) {
        int b = t / np;
        int p = idx[t];
        const float* s = xyz + ((long)b * n + p) * 3;
        nx[t * 3] = s[0]; nx[t * 3 + 1] = s[1]; nx[t * 3 + 2] = s[2];
    }
}

// ---------------------------------------------------------------------------
// 32 nearest neighbors per query row: distances into shared mem, 32 iterative
// argmin selections (lowest-index tie break; set matches stable argsort[:32]).
// ---------------------------------------------------------------------------
__global__ __launch_bounds__(256)
void k_knn(const float* __restrict__ nxyz, const float* __restrict__ xyz,
           int n, int np, int nsample, int* __restrict__ nn)
{
    __shared__ float sd[2048];
    __shared__ float rm[256];
    __shared__ int   ridx[256];
    __shared__ float src[3];
    const int r = blockIdx.x;       // b*np + s
    const int b = r / np;
    const int tid = threadIdx.x;
    const float* xb = xyz + (long)b * n * 3;
    if (tid == 0) { src[0] = nxyz[r * 3]; src[1] = nxyz[r * 3 + 1]; src[2] = nxyz[r * 3 + 2]; }
    __syncthreads();
    float ss = src[0] * src[0] + src[1] * src[1] + src[2] * src[2];
    for (int p = tid; p < n; p += 256) {
        float x = xb[p * 3], y = xb[p * 3 + 1], z = xb[p * 3 + 2];
        sd[p] = ss + x * x + y * y + z * z - 2.f * (src[0] * x + src[1] * y + src[2] * z);
    }
    __syncthreads();
    for (int it = 0; it < nsample; ++it) {
        float best = 3.4e38f; int bi = n;
        for (int p = tid; p < n; p += 256) {
            float d = sd[p];
            if (d < best || (d == best && p < bi)) { best = d; bi = p; }
        }
        rm[tid] = best; ridx[tid] = bi;
        __syncthreads();
        for (int s = 128; s > 0; s >>= 1) {
            if (tid < s) {
                if (rm[tid + s] < rm[tid] ||
                    (rm[tid + s] == rm[tid] && ridx[tid + s] < ridx[tid])) {
                    rm[tid] = rm[tid + s]; ridx[tid] = ridx[tid + s];
                }
            }
            __syncthreads();
        }
        if (tid == 0) { nn[(long)r * nsample + it] = ridx[0]; sd[ridx[0]] = 3.4e38f; }
        __syncthreads();
    }
}

// ---------------------------------------------------------------------------
// Build grouped features, transposed for GEMM: nf[g, d, j] with d in [0,2C):
//   d <  C : feat[b,d,nn[j]] - feat[b,d,center]   (centered)
//   d >= C : feat[b,d-C,center]                   (replicated center feature)
// ---------------------------------------------------------------------------
__global__ __launch_bounds__(256)
void k_group(const float* __restrict__ feat, const int* __restrict__ fps,
             const int* __restrict__ nn, float* __restrict__ nf,
             int Cf, int n, int np, int nsample)
{
    const int g = blockIdx.x;       // b*np + s
    const int b = g / np;
    const int cen = fps[g];
    const float* fb = feat + (long)b * Cf * n;
    float* og = nf + (long)g * (2 * Cf) * nsample;
    const int* nng = nn + (long)g * nsample;
    for (int i = threadIdx.x; i < 2 * Cf * nsample; i += 256) {
        int d = i / nsample, j = i % nsample;
        float v;
        if (d < Cf) v = fb[(long)d * n + nng[j]] - fb[(long)d * n + cen];
        else        v = fb[(long)(d - Cf) * n + cen];
        og[i] = v;
    }
}

// ---------------------------------------------------------------------------
// Row softmax with pre-scale (softmax(energy/factor)).
// ---------------------------------------------------------------------------
__global__ __launch_bounds__(256)
void k_softmax(float* __restrict__ a, int len, float scale)
{
    __shared__ float red[256];
    float* row = a + (long)blockIdx.x * len;
    int tid = threadIdx.x;
    float mx = -3.4e38f;
    for (int i = tid; i < len; i += 256) mx = fmaxf(mx, row[i] * scale);
    red[tid] = mx; __syncthreads();
    for (int s = 128; s > 0; s >>= 1) { if (tid < s) red[tid] = fmaxf(red[tid], red[tid + s]); __syncthreads(); }
    mx = red[0]; __syncthreads();
    float sum = 0.f;
    for (int i = tid; i < len; i += 256) { float e = expf(row[i] * scale - mx); row[i] = e; sum += e; }
    red[tid] = sum; __syncthreads();
    for (int s = 128; s > 0; s >>= 1) { if (tid < s) red[tid] += red[tid + s]; __syncthreads(); }
    float inv = 1.f / red[0];
    for (int i = tid; i < len; i += 256) row[i] *= inv;
}

// ---------------------------------------------------------------------------
// Host orchestration
// ---------------------------------------------------------------------------
struct BNP { const float *b, *g, *m, *v; };
struct SAP { const float *qk, *v, *vb; };
struct MAP { BNP bn1, bn2; SAP sa[4]; const float *w1, *w2; };
struct DAP { MAP c, p; };
struct LOP { BNP bn1, bn2; const float *w1, *w2; };

extern "C" void kernel_launch(void* const* d_in, const int* in_sizes, int n_in,
                              void* d_out, int out_size, void* d_ws, size_t ws_size,
                              hipStream_t stream)
{
    (void)out_size; (void)ws_size;
    const int Bc = 32, N1 = 2048, NS = 32;

    // Locate x (unique flat size 32*2048*3); params leaves follow canonical
    // JAX pytree order (dict keys sorted recursively).
    int base = 0;
    const float* x;
    if (in_sizes[0] == Bc * N1 * 3) { x = (const float*)d_in[0]; base = 1; }
    else                            { x = (const float*)d_in[n_in - 1]; base = 0; }

    int cur = base;
    auto nxt = [&]() { return (const float*)d_in[cur++]; };
    auto getBN = [&]() { BNP r; r.b = nxt(); r.g = nxt(); r.m = nxt(); r.v = nxt(); return r; };
    auto getSA = [&]() { SAP r; r.qk = nxt(); r.v = nxt(); r.vb = nxt(); return r; };
    auto getMAP = [&]() { MAP r; r.bn1 = getBN(); r.bn2 = getBN();
                          for (int i = 0; i < 4; ++i) r.sa[i] = getSA();
                          r.w1 = nxt(); r.w2 = nxt(); return r; };
    auto getDA = [&]() { DAP r; r.c = getMAP(); r.p = getMAP(); return r; };
    auto getLO = [&]() { LOP r; r.bn1 = getBN(); r.bn2 = getBN(); r.w1 = nxt(); r.w2 = nxt(); return r; };

    // sorted top-level param keys:
    const float* b_l2 = nxt();
    const float* b_l3 = nxt();
    BNP bn1p = getBN();
    BNP bn2p = getBN();
    const float* conv1 = nxt();
    const float* conv2 = nxt();
    DAP da1 = getDA(), da2 = getDA(), da3 = getDA();
    LOP lo0 = getLO(), lo1 = getLO(), lo2 = getLO();
    const float* w_l1 = nxt();
    const float* w_l2 = nxt();
    const float* w_l3 = nxt();

    // ---- workspace layout (bump allocator) ----
    char* ws = (char*)d_ws;
    size_t off = 0;
    auto alloc = [&](size_t bytes) {
        size_t o = (off + 255) & ~(size_t)255; off = o + bytes; return (float*)(ws + o);
    };
    float* h0     = alloc((size_t)Bc * 64 * N1 * 4);
    float* h1     = alloc((size_t)Bc * 64 * N1 * 4);
    float* arenaA = alloc((size_t)16384 * 128 * 32 * 4);   // nf tiles / attn
    float* arenaB = alloc((size_t)16384 * 128 * 32 * 4);   // GEMM temporaries
    float* feat1  = alloc((size_t)Bc * 128 * 512 * 4);
    float* dout1  = alloc((size_t)Bc * 128 * 512 * 4);
    float* feat2  = alloc((size_t)Bc * 256 * 128 * 4);
    float* dout2  = alloc((size_t)Bc * 256 * 128 * 4);
    float* feat3  = alloc((size_t)Bc * 1024 * 4);
    float* dout3  = alloc((size_t)Bc * 1024 * 4);
    float* nxyz1  = alloc((size_t)Bc * 512 * 3 * 4);
    float* nxyz2  = alloc((size_t)Bc * 128 * 3 * 4);
    float* nxyz3  = alloc((size_t)Bc * 1 * 3 * 4);
    int*   fps1   = (int*)alloc((size_t)Bc * 512 * 4);
    int*   fps2   = (int*)alloc((size_t)Bc * 128 * 4);
    int*   fps3   = (int*)alloc((size_t)Bc * 4);
    int*   nn1    = (int*)alloc((size_t)Bc * 512 * NS * 4);
    int*   nn2    = (int*)alloc((size_t)Bc * 128 * NS * 4);
    int*   nn3    = (int*)alloc((size_t)Bc * NS * 4);
    float* fcA    = alloc((size_t)Bc * 512 * 4);
    float* fcB    = alloc((size_t)Bc * 256 * 4);
    float* bnS    = alloc(1024 * 4);
    float* bnB    = alloc(1024 * 4);

    auto gemm = [&](const float* A, long lda, int tA,
                    const float* Bm, long ldb, int tB,
                    float* C, long ldc,
                    int M, int N, int K, long sA, long sB, long sC, int Z,
                    int epi, const float* e0, const float* e1,
                    float* pool, int poolN) {
        const int nw = (M % 64 == 0) ? 4 : 2;
        dim3 grid((N + 31) / 32, (M + nw * 16 - 1) / (nw * 16), Z);
        dim3 blk(nw * 32);
        if (nw == 4) {
            if (tA)      k_gemm<1, 0, 4><<<grid, blk, 0, stream>>>(A, lda, Bm, ldb, C, ldc, M, N, K, sA, sB, sC, epi, e0, e1, pool, poolN);
            else if (tB) k_gemm<0, 1, 4><<<grid, blk, 0, stream>>>(A, lda, Bm, ldb, C, ldc, M, N, K, sA, sB, sC, epi, e0, e1, pool, poolN);
            else         k_gemm<0, 0, 4><<<grid, blk, 0, stream>>>(A, lda, Bm, ldb, C, ldc, M, N, K, sA, sB, sC, epi, e0, e1, pool, poolN);
        } else {
            if (tA)      k_gemm<1, 0, 2><<<grid, blk, 0, stream>>>(A, lda, Bm, ldb, C, ldc, M, N, K, sA, sB, sC, epi, e0, e1, pool, poolN);
            else if (tB) k_gemm<0, 1, 2><<<grid, blk, 0, stream>>>(A, lda, Bm, ldb, C, ldc, M, N, K, sA, sB, sC, epi, e0, e1, pool, poolN);
            else         k_gemm<0, 0, 2><<<grid, blk, 0, stream>>>(A, lda, Bm, ldb, C, ldc, M, N, K, sA, sB, sC, epi, e0, e1, pool, poolN);
        }
    };
    auto bnprep = [&](const BNP& p, int C) {
        k_bnprep<<<(C + 255) / 256, 256, 0, stream>>>(p.g, p.b, p.m, p.v, bnS, bnB, C);
    };
    auto zero = [&](float* p, long n) {
        k_zero<<<(int)((n + 255) / 256), 256, 0, stream>>>(p, n);
    };

    // ---- sample_and_group stage ----
    auto runGroup = [&](const float* xyzS, const float* featS, int n, int Cf, int np,
                        int* fpsI, float* nx, int* nnI, float* nf) {
        k_fps<<<Bc, 256, 0, stream>>>(xyzS, n, np, fpsI);
        k_gather_xyz<<<(Bc * np + 255) / 256, 256, 0, stream>>>(xyzS, fpsI, nx, n, np, Bc * np);
        k_knn<<<Bc * np, 256, 0, stream>>>(nx, xyzS, n, np, NS, nnI);
        k_group<<<Bc * np, 256, 0, stream>>>(featS, fpsI, nnI, nf, Cf, n, np, NS);
    };

    // ---- local_op: 2 grouped GEMMs, max-over-samples fused as atomic pool ----
    auto runLO = [&](float* nf, const LOP& lp, int Din, int O1, int O2, int np, float* poolOut) {
        int G = Bc * np;
        bnprep(lp.bn1, O1);
        gemm(lp.w1, Din, 0, nf, NS, 0, arenaB, NS,
             O1, NS, Din, 0, (long)Din * NS, (long)O1 * NS, G,
             EPI_BNRELU, bnS, bnB, nullptr, 0);
        zero(poolOut, (long)Bc * O2 * np);
        bnprep(lp.bn2, O2);
        gemm(lp.w2, O1, 0, arenaB, NS, 0, nullptr, 0,
             O2, NS, O1, 0, (long)O1 * NS, 0, G,
             EPI_BNRELU_POOL, bnS, bnB, poolOut, np);
    };

    // ---- one multi_attn branch, accumulated into outAcc ----
    auto runMA = [&](const float* xin, const MAP& mp, int c, int n, float* outAcc) {
        int c4 = c / 4;
        float scale = 1.0f / sqrtf((float)c4);
        size_t cn = (size_t)Bc * c * n;
        float* x1 = arenaB;
        float* x2 = arenaB + cn;
        float* qa = x2 + cn;          // (B, 4, c4, n)
        float* va = qa + cn;
        float* cc = va + cn;          // concat (B, c, n)
        float* attn = arenaA;         // (B, 4, n, n)

        bnprep(mp.bn1, c);
        gemm(mp.w1, c, 0, xin, n, 0, x1, n, c, n, c, 0, (long)c * n, (long)c * n, Bc,
             EPI_BNRELU, bnS, bnB, nullptr, 0);
        bnprep(mp.bn2, c);
        gemm(mp.w2, c, 0, x1, n, 0, x2, n, c, n, c, 0, (long)c * n, (long)c * n, Bc,
             EPI_BNRELU, bnS, bnB, nullptr, 0);
        for (int h = 0; h < 4; ++h) {
            gemm(mp.sa[h].qk, c, 0, x2, n, 0, qa + (size_t)h * c4 * n, n,
                 c4, n, c, 0, (long)c * n, (long)c * n, Bc,
                 EPI_NONE, nullptr, nullptr, nullptr, 0);
            gemm(mp.sa[h].v, c, 0, x2, n, 0, va + (size_t)h * c4 * n, n,
                 c4, n, c, 0, (long)c * n, (long)c * n, Bc,
                 EPI_BIASROW, nullptr, mp.sa[h].vb, nullptr, 0);
        }
        // energy[n,m] = sum_c q[c,n] q[c,m]  -> transA GEMM, Z = B*4 heads
        gemm(qa, n, 1, qa, n, 0, attn, n,
             n, n, c4, (long)c4 * n, (long)c4 * n, (long)n * n, Bc * 4,
             EPI_NONE, nullptr, nullptr, nullptr, 0);
        k_softmax<<<Bc * 4 * n, 256, 0, stream>>>(attn, n, scale);
        // out[c4,m] = sum_n v[c4,n] attn[n,m]; head h writes at channel h*c4
        gemm(va, n, 0, attn, n, 0, cc, n,
             c4, n, n, (long)c4 * n, (long)n * n, (long)c4 * n, Bc * 4,
             EPI_NONE, nullptr, nullptr, nullptr, 0);
        k_add3<<<(int)((cn + 255) / 256), 256, 0, stream>>>(outAcc, cc, x2, (long)cn);
    };

    // ================= forward =================
    // initial MLP: (B,3,N) -> 64 -> 64 ; x is (B,N,3) row-major -> transB
    bnprep(bn1p, 64);
    gemm(conv1, 3, 0, x, 3, 1, h0, N1, 64, N1, 3, 0, (long)N1 * 3, (long)64 * N1, Bc,
         EPI_BNRELU, bnS, bnB, nullptr, 0);
    bnprep(bn2p, 64);
    gemm(conv2, 64, 0, h0, N1, 0, h1, N1, 64, N1, 64, 0, (long)64 * N1, (long)64 * N1, Bc,
         EPI_BNRELU, bnS, bnB, nullptr, 0);

    // stage 1: 2048 -> 512 points, C 64 -> 128
    runGroup(x, h1, N1, 64, 512, fps1, nxyz1, nn1, arenaA);
    runLO(arenaA, lo0, 128, 128, 128, 512, feat1);
    zero(dout1, (long)Bc * 128 * 512);
    runMA(feat1, da1.p, 128, 512, dout1);
    runMA(feat1, da1.c, 128, 512, dout1);

    // stage 2: 512 -> 128 points, C 128 -> 256
    runGroup(nxyz1, dout1, 512, 128, 128, fps2, nxyz2, nn2, arenaA);
    runLO(arenaA, lo1, 256, 256, 256, 128, feat2);
    zero(dout2, (long)Bc * 256 * 128);
    runMA(feat2, da2.p, 256, 128, dout2);
    runMA(feat2, da2.c, 256, 128, dout2);

    // stage 3: 128 -> 1 point, C 256 -> 1024
    runGroup(nxyz2, dout2, 128, 256, 1, fps3, nxyz3, nn3, arenaA);
    runLO(arenaA, lo2, 512, 1024, 1024, 1, feat3);
    zero(dout3, (long)Bc * 1024);
    runMA(feat3, da3.p, 1024, 1, dout3);
    runMA(feat3, da3.c, 1024, 1, dout3);

    // classifier head: (B,1024) @ W^T chains
    gemm(dout3, 1024, 0, w_l1, 1024, 1, fcA, 512, Bc, 512, 1024, 0, 0, 0, 1,
         EPI_RELU, nullptr, nullptr, nullptr, 0);
    gemm(fcA, 512, 0, w_l2, 512, 1, fcB, 256, Bc, 256, 512, 0, 0, 0, 1,
         EPI_BIASCOL_RELU, nullptr, b_l2, nullptr, 0);
    gemm(fcB, 256, 0, w_l3, 256, 1, (float*)d_out, 40, Bc, 40, 256, 0, 0, 0, 1,
         EPI_BIASCOL, nullptr, b_l3, nullptr, 0);
}
